// MergeModel_56100862820795
// MI455X (gfx1250) — matrix operand order repaired and checked by reading, not compiled, not run on hardware
//
#include <hip/hip_runtime.h>

// ---------------------------------------------------------------------------
// MI455X / gfx1250 MergeModel forward.
// Convs = implicit GEMM on v_wmma_f32_16x16x32_bf16, LDS-fed:
//   - packed A (weights) DMA'd into LDS via TENSOR_LOAD_TO_LDS (TDM)
//   - activation row strip staged in LDS per K-block, shared by 8 waves
// ---------------------------------------------------------------------------

typedef __attribute__((ext_vector_type(16))) __bf16 v16bf;
typedef __attribute__((ext_vector_type(8)))  float  v8f;
typedef unsigned int v4u __attribute__((ext_vector_type(4)));
typedef int          v8i __attribute__((ext_vector_type(8)));
typedef int          v4i __attribute__((ext_vector_type(4)));

#define HWFULL (512 * 512)

__device__ __forceinline__ unsigned short f2bf(float f) {
  unsigned int u = __float_as_uint(f);
  u += 0x7FFFu + ((u >> 16) & 1u);            // round-to-nearest-even
  return (unsigned short)(u >> 16);
}
__device__ __forceinline__ float bf2f(unsigned short h) {
  return __uint_as_float(((unsigned int)h) << 16);
}

// ---------------------------------------------------------------------------
// Repack conv weights (Cout,Cin,KH,KW) f32 -> WMMA A-fragment order, bf16.
// Lane 0-15 : M=lane, K 0-7 (v0-3), 16-23 (v4-7); lane 16-31: K 8-15 / 24-31.
// Element q: kk = (q<8 ? q : q+8) + hi*8 ; dst = ((mt*KB+kb)*32 + lane)*16 + q
// ---------------------------------------------------------------------------
__global__ void repack_a(const float* __restrict__ w, unsigned short* __restrict__ dst,
                         int Cout, int Cin, int KH, int KW, int Ktot, int KB, int Mtiles) {
  int idx = blockIdx.x * blockDim.x + threadIdx.x;
  int total = Mtiles * KB * 32 * 16;
  if (idx >= total) return;
  int q    = idx & 15;
  int lane = (idx >> 4) & 31;
  int kb   = (idx >> 9) % KB;
  int mt   = idx / (KB << 9);
  int lo = lane & 15, hi = lane >> 4;
  int kk = (q < 8 ? q : q + 8) + hi * 8;
  int k  = kb * 32 + kk;
  int m  = mt * 16 + lo;
  unsigned short v = 0;
  if (k < Ktot && m < Cout) {
    int khkw = KH * KW;
    int c    = k / khkw;
    int rem  = k - c * khkw;
    v = f2bf(w[((size_t)m * Cin + c) * khkw + rem]);
  }
  dst[idx] = v;
}

// ---------------------------------------------------------------------------
// Implicit-GEMM convolution, LDS-fed WMMA.
// Workgroup = 8 waves -> one (mt, row y, 128-pixel strip).
// ---------------------------------------------------------------------------
#define KMAX   1408
#define KBMAX  44
#define ROWMAX 32
#define WSP    144
__global__ void conv_wmma(const unsigned short* __restrict__ act,
                          const unsigned short* __restrict__ apack,
                          const float* __restrict__ bias,
                          unsigned short* __restrict__ out,
                          int Cout, int Cin, int H, int W,
                          int KH, int KW, int dil, int pad,
                          int Ktot, int KB, int Mtiles, int do_relu) {
  __shared__ unsigned short sA[KBMAX * 512];     // packed A for this M-tile
  __shared__ unsigned short sIn[ROWMAX * WSP];   // staged input strip rows
  __shared__ short s_rabs[KMAX];                 // k -> c*KH + ki
  __shared__ short s_ox[KMAX];                   // k -> dil*kj
  __shared__ short s_c0[64];                     // kb -> first channel
  __shared__ short s_c1[64];                     // kb -> last channel

  int SX  = (W + 127) >> 7;
  int bid = blockIdx.x;
  int mt  = bid / (H * SX);
  int rem = bid - mt * (H * SX);
  int y   = rem / SX;
  int x0s = (rem - y * SX) << 7;

  // ---- async DMA of packed A into LDS via the Tensor Data Mover ----
#if __has_builtin(__builtin_amdgcn_tensor_load_to_lds)
  if ((threadIdx.x >> 5) == 0) {
    unsigned long long ga =
        (unsigned long long)(size_t)(apack + (size_t)mt * KB * 512);
    unsigned nd   = (unsigned)(KB * 256);        // dwords to move
    unsigned lofs = (unsigned)(size_t)&sA[0];    // LDS byte offset (low 32 bits)
    v4u g0 = { 1u,                               // count=1, user mode
               lofs,
               (unsigned)(ga & 0xFFFFFFFFu),
               (unsigned)((ga >> 32) & 0x01FFFFFFu) | (2u << 30) }; // type=2
    v8i g1 = { (int)(2u << 16),                          // data_size = 4B
               (int)((nd & 0xFFFFu) << 16),              // tensor_dim0 lo
               (int)(((nd >> 16) & 0xFFFFu) | (1u << 16)), // dim0 hi | dim1=1
               (int)(nd << 16),                          // tile_dim0 = nd
               (int)1,                                   // tile_dim1 = 1
               (int)nd,                                  // dim0_stride lo
               0, 0 };
    v4i z4 = { 0, 0, 0, 0 };
#if defined(__clang_major__) && __clang_major__ >= 23
    v8i z8 = { 0, 0, 0, 0, 0, 0, 0, 0 };
    __builtin_amdgcn_tensor_load_to_lds(g0, g1, z4, z4, z8, 0);
#else
    __builtin_amdgcn_tensor_load_to_lds(g0, g1, z4, z4, 0);
#endif
  }
#else
  for (int i = threadIdx.x; i < KB * 512; i += blockDim.x)
    sA[i] = apack[(size_t)mt * KB * 512 + i];
#endif

  // ---- im2col lookup tables ----
  int khkw = KH * KW;
  for (int k = threadIdx.x; k < Ktot; k += blockDim.x) {
    int c   = k / khkw;
    int r   = k - c * khkw;
    int ki  = r / KW;
    int kj  = r - ki * KW;
    s_rabs[k] = (short)(c * KH + ki);
    s_ox[k]   = (short)(dil * kj);
  }
  for (int kb = threadIdx.x; kb < KB; kb += blockDim.x) {
    int c0 = (kb * 32) / khkw;
    int c1 = (kb * 32 + 31) / khkw;
    if (c1 > Cin - 1) c1 = Cin - 1;
    s_c0[kb] = (short)c0;
    s_c1[kb] = (short)c1;
  }

#if __has_builtin(__builtin_amdgcn_tensor_load_to_lds)
  if ((threadIdx.x >> 5) == 0) __builtin_amdgcn_s_wait_tensorcnt(0);
#endif
  __syncthreads();

  int lane = threadIdx.x & 31;
  int wti  = threadIdx.x >> 5;                   // wave's 16-pixel sub-strip
  int lo   = lane & 15, hi = lane >> 4;
  int xoff = (wti << 4) + lo;                    // col base inside strip
  int x    = x0s + xoff;
  int WS   = 128 + dil * (KW - 1);               // staged strip width (<=140)

  v8f acc = {};
  for (int kb = 0; kb < KB; ++kb) {
    int c0    = s_c0[kb];
    int nrows = (s_c1[kb] - c0 + 1) * KH;
    // ---- stage the needed input rows (bounds baked in as zeros) ----
    for (int i = threadIdx.x; i < nrows * WS; i += blockDim.x) {
      int lr  = i / WS;
      int col = i - lr * WS;
      int cc  = c0 + lr / KH;
      int ky  = lr - (lr / KH) * KH;
      int gy  = y + dil * ky - pad;
      int gx  = x0s - pad + col;
      unsigned short v = 0;
      if (gy >= 0 && gy < H && gx >= 0 && gx < W)
        v = act[(size_t)cc * H * W + (size_t)gy * W + gx];
      sIn[lr * WSP + col] = v;
    }
    __syncthreads();

    // ---- A fragment from LDS ----
    v16bf av;
    __builtin_memcpy(&av, &sA[kb * 512 + lane * 16], 32);   // 2x ds_load_b128

    // ---- B fragment from staged strip (lane lo = N, K = (q<8 ? q : q+8) + hi*8) ----
    union { v16bf v; unsigned short s[16]; } B;
    int kbase = kb * 32 + hi * 8;
    int rbase = c0 * KH;
#pragma unroll
    for (int q = 0; q < 16; ++q) {
      int k = kbase + (q < 8 ? q : q + 8);
      unsigned short bv = 0;
      if (k < Ktot) {
        int lr  = s_rabs[k] - rbase;
        int col = xoff + s_ox[k];
        bv = sIn[lr * WSP + col];
      }
      B.s[q] = bv;
    }
    acc = __builtin_amdgcn_wmma_f32_16x16x32_bf16(false, av, false, B.v,
                                                  (short)0, acc, false, false);
    __syncthreads();
  }

  if (x < W) {
#pragma unroll
    for (int r = 0; r < 8; ++r) {
      int m = mt * 16 + r + 8 * hi;              // C/D: M = r + 8*hi
      if (m < Cout) {
        float v = acc[r] + bias[m];
        if (do_relu) v = fmaxf(v, 0.0f);
        out[(size_t)m * H * W + (size_t)y * W + x] = f2bf(v);
      }
    }
  }
}

// ---------------------------------------------------------------------------
// Elementwise / pooling helpers
// ---------------------------------------------------------------------------
__global__ void cast_f32_bf16(const float* __restrict__ in, unsigned short* __restrict__ out, int n) {
  int i = blockIdx.x * blockDim.x + threadIdx.x;
  if (i < n) out[i] = f2bf(in[i]);
}

__global__ void zero_f32(float* p, int n) {
  int i = blockIdx.x * blockDim.x + threadIdx.x;
  if (i < n) p[i] = 0.0f;
}

__global__ void maxpool2(const unsigned short* __restrict__ in, unsigned short* __restrict__ out,
                         int C, int H, int W, int pad, int Ho, int Wo) {
  int i = blockIdx.x * blockDim.x + threadIdx.x;
  int tot = C * Ho * Wo;
  if (i >= tot) return;
  int wo = i % Wo;
  int t  = i / Wo;
  int ho = t % Ho;
  int c  = t / Ho;
  float m = -3.4e38f;
#pragma unroll
  for (int dy = 0; dy < 2; ++dy)
#pragma unroll
    for (int dx = 0; dx < 2; ++dx) {
      int iy = ho + dy - pad, ix = wo + dx - pad;
      if (iy >= 0 && iy < H && ix >= 0 && ix < W)
        m = fmaxf(m, bf2f(in[(size_t)c * H * W + (size_t)iy * W + ix]));
    }
  out[i] = f2bf(m);
}

__global__ void count_ids(const int* __restrict__ rid, const int* __restrict__ cid,
                          float* rc, float* cc, int H, int W) {
  int i = blockIdx.x * blockDim.x + threadIdx.x;
  if (i < H) atomicAdd(&rc[rid[i]], 1.0f);
  if (i < W) atomicAdd(&cc[cid[i]], 1.0f);
}

__global__ void inv_area_k(const float* rc, const float* cc, float* ia) {
  int i = blockIdx.x * blockDim.x + threadIdx.x;
  if (i < 600) ia[i] = 1.0f / (rc[i / 20] * cc[i % 20]);
}

__global__ void cell_pool(const unsigned short* __restrict__ t,
                          const int* __restrict__ rid, const int* __restrict__ cid,
                          float* cs, int C, int do_sigmoid) {
  int i = blockIdx.x * blockDim.x + threadIdx.x;
  if (i >= C * HWFULL) return;
  int c   = i / HWFULL;
  int rem = i - c * HWFULL;
  int y   = rem >> 9;
  int x   = rem & 511;
  float v = bf2f(t[i]);
  if (do_sigmoid) v = 1.0f / (1.0f + expf(-v));
  atomicAdd(&cs[c * 600 + rid[y] * 20 + cid[x]], v);
}

__global__ void broadcast_cells(const float* __restrict__ cs, const float* __restrict__ ia,
                                const int* __restrict__ rid, const int* __restrict__ cid,
                                unsigned short* __restrict__ out, int C) {
  int i = blockIdx.x * blockDim.x + threadIdx.x;
  if (i >= C * HWFULL) return;
  int c   = i / HWFULL;
  int rem = i - c * HWFULL;
  int y   = rem >> 9;
  int x   = rem & 511;
  int g   = rid[y] * 20 + cid[x];
  out[i]  = f2bf(cs[c * 600 + g] * ia[g]);
}

__global__ void save_pools(const float* cs, const float* ia, float* p) {
  int i = blockIdx.x * blockDim.x + threadIdx.x;
  if (i < 600) p[i] = cs[i] * ia[i];
}

// P: up1,up2,dn1,dn2,lf1,lf2,rt1,rt2 (each 30x20)
// out: D1(29x20) D2(29x20) R1(30x19) R2(30x19) -> 2300 floats
__global__ void combine_out(const float* __restrict__ P, float* __restrict__ out) {
  int i = blockIdx.x * blockDim.x + threadIdx.x;
  if (i >= 2300) return;
  float a2, b2;
  if (i < 1160) {
    const float* a = (i < 580) ? P : P + 600;
    const float* b = (i < 580) ? P + 1200 : P + 1800;
    int j = i % 580, r = j / 20, k = j % 20;
    a2 = a[(r + 1) * 20 + k];
    b2 = b[r * 20 + k];
  } else {
    int j = i - 1160;
    const float* a = (j < 570) ? P + 2400 : P + 3000;
    const float* b = (j < 570) ? P + 3600 : P + 4200;
    int jj = j % 570, r = jj / 19, k = jj % 19;
    a2 = a[r * 20 + k + 1];
    b2 = b[r * 20 + k];
  }
  out[i] = 0.5f * (a2 * b2) + 0.25f * (a2 + b2);
}

// ---------------------------------------------------------------------------
// Host orchestration
// ---------------------------------------------------------------------------
static inline int cdiv(long a, long b) { return (int)((a + b - 1) / b); }

static void run_conv(hipStream_t st, const float* w, const float* b,
                     const unsigned short* in, unsigned short* out,
                     unsigned short* apack,
                     int Cout, int Cin, int KH, int KW, int dil, int pad,
                     int H, int W, int relu) {
  int Ktot = Cin * KH * KW;
  int KB   = (Ktot + 31) / 32;
  int Mt   = (Cout + 15) / 16;
  int n    = Mt * KB * 512;
  repack_a<<<cdiv(n, 256), 256, 0, st>>>(w, apack, Cout, Cin, KH, KW, Ktot, KB, Mt);
  int SX = (W + 127) / 128;
  conv_wmma<<<H * SX * Mt, 256, 0, st>>>(in, apack, b, out, Cout, Cin, H, W,
                                         KH, KW, dil, pad, Ktot, KB, Mt, relu);
}

static void run_block(hipStream_t st, void* const* d_in, int wbase,
                      const int* rid, const int* cid,
                      unsigned short* X, unsigned short* XN, unsigned short* TOPB,
                      unsigned short* BOT, unsigned short* APACK,
                      float* CS_TOP, float* CS_BOT, float* IA, float* poolsDst,
                      int cin, int cout, int build_out) {
  const float* wd1 = (const float*)d_in[wbase + 0];
  const float* bd1 = (const float*)d_in[wbase + 1];
  const float* wd2 = (const float*)d_in[wbase + 2];
  const float* bd2 = (const float*)d_in[wbase + 3];
  const float* wd3 = (const float*)d_in[wbase + 4];
  const float* bd3 = (const float*)d_in[wbase + 5];
  const float* wt  = (const float*)d_in[wbase + 6];
  const float* bt  = (const float*)d_in[wbase + 7];
  const float* wb  = (const float*)d_in[wbase + 8];
  const float* bb  = (const float*)d_in[wbase + 9];

  unsigned short* feat = build_out ? (XN + (size_t)cout * HWFULL) : TOPB;
  run_conv(st, wd1, bd1, X, feat,                       APACK, 6, cin, 5, 5, 1, 2, 512, 512, 1);
  run_conv(st, wd2, bd2, X, feat + 6 * (size_t)HWFULL,  APACK, 6, cin, 5, 5, 2, 4, 512, 512, 1);
  run_conv(st, wd3, bd3, X, feat + 12 * (size_t)HWFULL, APACK, 6, cin, 5, 5, 3, 6, 512, 512, 1);

  if (build_out) {
    run_conv(st, wt, bt, feat, TOPB, APACK, cout, 18, 1, 1, 1, 0, 512, 512, 0);
    zero_f32<<<cdiv(cout * 600, 256), 256, 0, st>>>(CS_TOP, cout * 600);
    cell_pool<<<cdiv((long)cout * HWFULL, 256), 256, 0, st>>>(TOPB, rid, cid, CS_TOP, cout, 0);
    broadcast_cells<<<cdiv((long)cout * HWFULL, 256), 256, 0, st>>>(CS_TOP, IA, rid, cid, XN, cout);
  }

  run_conv(st, wb, bb, feat, BOT, APACK, 1, 18, 1, 1, 1, 0, 512, 512, 0);
  zero_f32<<<3, 256, 0, st>>>(CS_BOT, 600);
  cell_pool<<<cdiv(HWFULL, 256), 256, 0, st>>>(BOT, rid, cid, CS_BOT, 1, 1);
  if (build_out)
    broadcast_cells<<<cdiv(HWFULL, 256), 256, 0, st>>>(CS_BOT, IA, rid, cid,
                                                       XN + (size_t)(cout + 18) * HWFULL, 1);
  if (poolsDst) save_pools<<<3, 256, 0, st>>>(CS_BOT, IA, poolsDst);
}

extern "C" void kernel_launch(void* const* d_in, const int* in_sizes, int n_in,
                              void* d_out, int out_size, void* d_ws, size_t ws_size,
                              hipStream_t stream) {
  (void)in_sizes; (void)n_in; (void)out_size; (void)ws_size;
  const float* x   = (const float*)d_in[0];
  const int*   rid = (const int*)d_in[1];
  const int*   cid = (const int*)d_in[2];

  unsigned short* BIGA  = (unsigned short*)d_ws;           // 55*HW
  unsigned short* BIGB  = BIGA + 14417920;                 // 55*HW
  unsigned short* BIGC  = BIGB + 14417920;                 // 36*HW
  unsigned short* FEATS = BIGC + 9437184;                  // 32*HW
  unsigned short* BOT   = FEATS + 8388608;                 // HW
  unsigned short* APACK = BOT + 262144;                    // 65536
  float* CS_TOP = (float*)(APACK + 65536);                 // 64*600
  float* CS_BOT = CS_TOP + 38400;                          // 600
  float* IA     = CS_BOT + 600;                            // 600
  float* RC     = IA + 600;                                // 30
  float* CC     = RC + 30;                                 // 20
  float* P      = CC + 20;                                 // 8*600

  zero_f32<<<1, 64, 0, stream>>>(RC, 50);
  count_ids<<<2, 256, 0, stream>>>(rid, cid, RC, CC, 512, 512);
  inv_area_k<<<3, 256, 0, stream>>>(RC, CC, IA);

  cast_f32_bf16<<<cdiv(8L * HWFULL, 256), 256, 0, stream>>>(x, BIGA, 8 * HWFULL);
  const float* w1 = (const float*)d_in[3];  const float* b1 = (const float*)d_in[4];
  const float* w2 = (const float*)d_in[5];  const float* b2 = (const float*)d_in[6];
  const float* w4 = (const float*)d_in[7];  const float* b4 = (const float*)d_in[8];
  const float* w5 = (const float*)d_in[9];  const float* b5 = (const float*)d_in[10];
  run_conv(stream, w1, b1, BIGA, BIGB, APACK, 18, 8, 7, 7, 1, 3, 512, 512, 1);
  run_conv(stream, w2, b2, BIGB, BIGA, APACK, 18, 18, 7, 7, 1, 3, 512, 512, 1);
  maxpool2<<<cdiv(18L * 513 * 513, 256), 256, 0, stream>>>(BIGA, BIGB, 18, 512, 512, 1, 513, 513);
  run_conv(stream, w4, b4, BIGB, BIGA, APACK, 18, 18, 5, 5, 1, 2, 513, 513, 1);
  run_conv(stream, w5, b5, BIGA, BIGB, APACK, 32, 18, 5, 5, 1, 2, 513, 513, 1);
  maxpool2<<<cdiv(32L * HWFULL, 256), 256, 0, stream>>>(BIGB, FEATS, 32, 513, 513, 0, 512, 512);

  for (int br = 0; br < 4; ++br) {
    int base = 11 + br * 30;
    run_block(stream, d_in, base +  0, rid, cid, FEATS, BIGB, BIGC, BOT, APACK,
              CS_TOP, CS_BOT, IA, nullptr,             32, 36, 1);
    run_block(stream, d_in, base + 10, rid, cid, BIGB,  BIGA, BIGC, BOT, APACK,
              CS_TOP, CS_BOT, IA, P + br * 1200,       55, 36, 1);
    run_block(stream, d_in, base + 20, rid, cid, BIGA,  BIGA, BIGC, BOT, APACK,
              CS_TOP, CS_BOT, IA, P + br * 1200 + 600, 55, 64, 0);
  }

  combine_out<<<cdiv(2300, 256), 256, 0, stream>>>(P, (float*)d_out);
}
